// LDA_AQU_4415226380824
// MI455X (gfx1250) — matrix operand order repaired and checked by reading, not compiled, not run on hardware
//
#include <hip/hip_runtime.h>
#include <math.h>

typedef __attribute__((ext_vector_type(16))) _Float16 v16h;
typedef __attribute__((ext_vector_type(8)))  _Float16 v8h;
typedef __attribute__((ext_vector_type(8)))  float    v8f;

#define CIN   128
#define HIDC  32
#define NHD   4
#define NGRP  2
#define GCH   16
#define HDK   8
#define HIN   64
#define WIN   64
#define OHW   128
#define N1    (OHW*OHW)     /* 16384 */
#define HW    (HIN*WIN)     /* 4096  */
#define BSZ   4
#define SCALE_Q   0.35355339059327379f
#define SRC_SCALE (63.0f/127.0f)

// ---------------- weight convert: Wh[64][128] f16, rows 0..31=wq, 32..63=wk ---
__global__ void k_wcvt(const float* __restrict__ wq, const float* __restrict__ wk,
                       _Float16* __restrict__ Wh) {
  int i = blockIdx.x * 256 + threadIdx.x;
  if (i >= 64 * 128) return;
  int row = i >> 7, col = i & 127;
  float v = (row < 32) ? wq[row * 128 + col] : wk[(row - 32) * 128 + col];
  Wh[i] = (_Float16)v;
}

// ---------------- channel LayerNorm -> f16 staging xnh[p][128] ----------------
__global__ void k_ln(const float* __restrict__ x, const float* __restrict__ lw,
                     const float* __restrict__ lb, _Float16* __restrict__ xnh) {
  int tid  = threadIdx.x;
  int lane = tid & 31;
  int p    = blockIdx.x * 8 + (tid >> 5);     // pixel index, b*4096+hw
  int b = p >> 12, hw = p & 4095;
  const float* base = x + (size_t)b * CIN * HW + hw;
  int c0 = lane * 4;
  float v[4]; float s = 0.f, sq = 0.f;
  #pragma unroll
  for (int i = 0; i < 4; i++) { v[i] = base[(size_t)(c0 + i) * HW]; s += v[i]; sq += v[i] * v[i]; }
  #pragma unroll
  for (int m = 1; m < 32; m <<= 1) { s += __shfl_xor(s, m, 32); sq += __shfl_xor(sq, m, 32); }
  float mean = s * (1.f / 128.f);
  float var  = sq * (1.f / 128.f) - mean * mean;
  float rstd = rsqrtf(var + 1e-6f);
  #pragma unroll
  for (int i = 0; i < 4; i++) {
    int c = c0 + i;
    float xn = lw[c] * ((v[i] - mean) * rstd) + lb[c];
    xnh[(size_t)p * 128 + c] = (_Float16)xn;
  }
}

// ---------------- q,k projection GEMM via v_wmma_f32_16x16x32_f16 -------------
// D[16 out-ch x 16 pixels] tiles; A = weights (M=ch,K), B = xn (K, N=pixels).
__global__ void k_gemm(const _Float16* __restrict__ xnh, const _Float16* __restrict__ Wh,
                       float* __restrict__ qbuf, float* __restrict__ kbuf) {
  int tid  = blockIdx.x * 256 + threadIdx.x;
  int wave = tid >> 5;
  int lane = tid & 31;
  int p0   = wave * 16;                    // 16-pixel tile
  int hi   = (lane & 16) ? 1 : 0;

  // B fragments: lane holds K-contiguous 16 halfs of pixel (p0 + lane%16)
  int pl = p0 + (lane & 15);
  const _Float16* xr = xnh + (size_t)pl * 128 + (hi ? 16 : 0);
  v16h bfrag[4];
  #pragma unroll
  for (int kt = 0; kt < 4; kt++) bfrag[kt] = *(const v16h*)(xr + kt * 32);

  int pix = p0 + (lane & 15);
  int b = pix >> 12, hw = pix & 4095;

  #pragma unroll
  for (int mt = 0; mt < 4; mt++) {         // 64 output channels = q(32) || k(32)
    int m0 = mt * 16;
    v8f acc = {0.f, 0.f, 0.f, 0.f, 0.f, 0.f, 0.f, 0.f};
    int row = m0 + (lane & 15);
    const _Float16* wr = Wh + row * 128 + (hi ? 16 : 0);
    #pragma unroll
    for (int kt = 0; kt < 4; kt++) {
      v8h lo = *(const v8h*)(wr + kt * 32);
      v8h hh = *(const v8h*)(wr + kt * 32 + 8);
      v16h a;
      #pragma unroll
      for (int i = 0; i < 8; i++) { a[i] = lo[i]; a[8 + i] = hh[i]; }
      acc = __builtin_amdgcn_wmma_f32_16x16x32_f16(false, a, false, bfrag[kt],
                                                   (short)0, acc, false, false);
    }
    #pragma unroll
    for (int r = 0; r < 8; r++) {
      int ng = m0 + r + (hi ? 8 : 0);      // global out channel (0..63)
      float val = acc[r];
      if (ng < 32) qbuf[((size_t)b * 32 + ng) * HW + hw] = val;
      else         kbuf[((size_t)b * 32 + (ng - 32)) * HW + hw] = val;
    }
  }
}

// ---------------- bilinear 2x upsample of q (align_corners) -------------------
__global__ void k_upsample(const float* __restrict__ qbuf, float* __restrict__ qhi) {
  int idx = blockIdx.x * 256 + threadIdx.x;       // B*32*16384
  int n  = idx & (N1 - 1);
  int ch = (idx >> 14) & 31;
  int b  = idx >> 19;
  int y = n >> 7, xx = n & 127;
  float sy = y * SRC_SCALE, sx = xx * SRC_SCALE;
  int y0 = (int)sy, x0 = (int)sx;
  int y1 = min(y0 + 1, 63), x1 = min(x0 + 1, 63);
  float wy = sy - (float)y0, wx = sx - (float)x0;
  const float* src = qbuf + ((size_t)b * 32 + ch) * HW;
  float v00 = src[y0 * 64 + x0], v01 = src[y0 * 64 + x1];
  float v10 = src[y1 * 64 + x0], v11 = src[y1 * 64 + x1];
  float t0 = v00 + (v01 - v00) * wx;
  float t1 = v10 + (v11 - v10) * wx;
  qhi[idx] = t0 + (t1 - t0) * wy;
}

// ---------------- depthwise 3x3 + channel-LN(16) + SiLU ----------------------
__global__ void k_dwln(const float* __restrict__ qhi, const float* __restrict__ dw,
                       const float* __restrict__ ow_, const float* __restrict__ ob_,
                       float* __restrict__ tbuf) {
  int idx = blockIdx.x * 256 + threadIdx.x;       // 8*16384
  int n  = idx & (N1 - 1);
  int bg = idx >> 14;
  int b = bg >> 1, g = bg & 1;
  int y = n >> 7, xx = n & 127;
  float tl[16]; float s = 0.f, sq = 0.f;
  for (int cg = 0; cg < 16; cg++) {
    const float* src = qhi + ((size_t)b * 32 + g * 16 + cg) * N1;
    float acc = 0.f;
    #pragma unroll
    for (int kh = 0; kh < 3; kh++) {
      int yy = y + kh - 1; if (yy < 0 || yy > 127) continue;
      #pragma unroll
      for (int kw = 0; kw < 3; kw++) {
        int xc = xx + kw - 1; if (xc < 0 || xc > 127) continue;
        acc += src[yy * 128 + xc] * dw[cg * 9 + kh * 3 + kw];
      }
    }
    tl[cg] = acc; s += acc; sq += acc * acc;
  }
  float mean = s * (1.f / 16.f);
  float var  = sq * (1.f / 16.f) - mean * mean;
  float rstd = rsqrtf(var + 1e-6f);
  for (int cg = 0; cg < 16; cg++) {
    float u = ow_[cg] * ((tl[cg] - mean) * rstd) + ob_[cg];
    float sig = 1.f / (1.f + __expf(-u));
    tbuf[((size_t)bg * 16 + cg) * N1 + n] = u * sig;
  }
}

// ---------------- 3x3 conv 16->18 + tanh*RANGE + base + pixel index ----------
// Emits absolute source-image sampling coords (iy,ix) per (bg,tap,pixel).
__global__ void k_offsets(const float* __restrict__ tbuf, const float* __restrict__ offw,
                          const float* __restrict__ offb, float* __restrict__ coords) {
  int idx = blockIdx.x * 256 + threadIdx.x;       // 8*16384
  int n  = idx & (N1 - 1);
  int bg = idx >> 14;
  int y = n >> 7, xx = n & 127;
  float acc[18];
  #pragma unroll
  for (int co = 0; co < 18; co++) acc[co] = offb[co];
  for (int ci = 0; ci < 16; ci++) {
    const float* src = tbuf + ((size_t)bg * 16 + ci) * N1;
    float tv[9];
    #pragma unroll
    for (int kh = 0; kh < 3; kh++)
      #pragma unroll
      for (int kw = 0; kw < 3; kw++) {
        int yy = y + kh - 1, xc = xx + kw - 1;
        tv[kh * 3 + kw] = (yy >= 0 && yy < 128 && xc >= 0 && xc < 128)
                            ? src[yy * 128 + xc] : 0.f;
      }
    for (int co = 0; co < 18; co++) {
      const float* w = offw + ((size_t)co * 16 + ci) * 9;
      float a = 0.f;
      #pragma unroll
      for (int kk = 0; kk < 9; kk++) a += tv[kk] * w[kk];
      acc[co] += a;
    }
  }
  const float br[9] = {-1,-1,-1, 0,0,0, 1,1,1};
  const float bc[9] = {-1, 0, 1,-1,0,1,-1,0,1};
  #pragma unroll
  for (int tap = 0; tap < 9; tap++) {
    float rr = tanhf(acc[2 * tap    ]) * 11.f + br[tap] + (float)y;
    float cc = tanhf(acc[2 * tap + 1]) * 11.f + bc[tap] + (float)xx;
    float* cp = coords + (((size_t)bg * 9 + tap) * N1 + n) * 2;
    cp[0] = rr * SRC_SCALE;   // iy in 64x64 source coords
    cp[1] = cc * SRC_SCALE;   // ix
  }
}

// ---------------- fused gather + 9-way softmax attention (wave per pixel) -----
__global__ void k_attn(const float* __restrict__ x, const float* __restrict__ kbuf,
                       const float* __restrict__ qhi, const float* __restrict__ coords,
                       const float* __restrict__ rpb, float* __restrict__ out) {
  int tid  = blockIdx.x * 256 + threadIdx.x;
  int wave = tid >> 5, lane = tid & 31;
  int pix = wave;                         // 0..65535
  int b = pix >> 14, n = pix & 16383;

  // lanes 0..17 hold the 18 (group,tap) sample coords for this pixel
  float my_iy = 0.f, my_ix = 0.f;
  if (lane < 18) {
    int g = lane / 9, tap = lane - g * 9;
    const float* cp = coords + (((size_t)(b * 2 + g) * 9 + tap) * N1 + n) * 2;
    my_iy = cp[0]; my_ix = cp[1];
  }
  int h = lane >> 3, cq = lane & 7, gl = lane >> 4;   // head, qk-dim, group
  float qv = qhi[((size_t)b * 32 + h * 8 + cq) * N1 + n] * SCALE_Q;
  const float* kc = kbuf + ((size_t)b * 32 + gl * 16 + (h & 1) * 8 + cq) * HW;

  float iys[9], ixs[9], logit[9];
  #pragma unroll
  for (int tap = 0; tap < 9; tap++) {
    int src = gl * 9 + tap;
    float iy = __shfl(my_iy, src, 32);
    float ix = __shfl(my_ix, src, 32);
    iys[tap] = iy; ixs[tap] = ix;
    float y0f = floorf(iy), x0f = floorf(ix);
    int y0 = (int)y0f, x0 = (int)x0f;
    float wy = iy - y0f, wx = ix - x0f;
    bool yv0 = (y0 >= 0 && y0 < 64),  yv1 = (y0 >= -1 && y0 < 63);
    bool xv0 = (x0 >= 0 && x0 < 64),  xv1 = (x0 >= -1 && x0 < 63);
    float v00 = (yv0 && xv0) ? kc[y0 * 64 + x0]           : 0.f;
    float v01 = (yv0 && xv1) ? kc[y0 * 64 + x0 + 1]       : 0.f;
    float v10 = (yv1 && xv0) ? kc[(y0 + 1) * 64 + x0]     : 0.f;
    float v11 = (yv1 && xv1) ? kc[(y0 + 1) * 64 + x0 + 1] : 0.f;
    float kv = v00 * (1.f - wy) * (1.f - wx) + v01 * (1.f - wy) * wx
             + v10 * wy * (1.f - wx)         + v11 * wy * wx;
    float part = qv * (kv + rpb[(h * 9 + tap) * 8 + cq]);
    part += __shfl_xor(part, 1, 32);
    part += __shfl_xor(part, 2, 32);
    part += __shfl_xor(part, 4, 32);    // dot over 8 q/k dims
    logit[tap] = part;
  }
  float m = logit[0];
  #pragma unroll
  for (int tp = 1; tp < 9; tp++) m = fmaxf(m, logit[tp]);
  float attn[9]; float ssum = 0.f;
  #pragma unroll
  for (int tp = 0; tp < 9; tp++) { attn[tp] = __expf(logit[tp] - m); ssum += attn[tp]; }
  float inv = 1.f / ssum;

  // v phase: each lane accumulates 4 output channels of its head
  int cv0 = (lane & 7) * 4;
  const float* vc = x + ((size_t)b * 128 + gl * 64 + (h & 1) * 32 + cv0) * HW;
  __builtin_prefetch(vc, 0, 1);
  float acc[4] = {0.f, 0.f, 0.f, 0.f};
  #pragma unroll
  for (int tap = 0; tap < 9; tap++) {
    float a = attn[tap] * inv;
    float iy = iys[tap], ix = ixs[tap];
    float y0f = floorf(iy), x0f = floorf(ix);
    int y0 = (int)y0f, x0 = (int)x0f;
    float wy = iy - y0f, wx = ix - x0f;
    bool yv0 = (y0 >= 0 && y0 < 64),  yv1 = (y0 >= -1 && y0 < 63);
    bool xv0 = (x0 >= 0 && x0 < 64),  xv1 = (x0 >= -1 && x0 < 63);
    float w00 = (1.f - wy) * (1.f - wx) * a, w01 = (1.f - wy) * wx * a;
    float w10 = wy * (1.f - wx) * a,         w11 = wy * wx * a;
    #pragma unroll
    for (int c = 0; c < 4; c++) {
      const float* p = vc + (size_t)c * HW;
      float v00 = (yv0 && xv0) ? p[y0 * 64 + x0]           : 0.f;
      float v01 = (yv0 && xv1) ? p[y0 * 64 + x0 + 1]       : 0.f;
      float v10 = (yv1 && xv0) ? p[(y0 + 1) * 64 + x0]     : 0.f;
      float v11 = (yv1 && xv1) ? p[(y0 + 1) * 64 + x0 + 1] : 0.f;
      acc[c] += v00 * w00 + v01 * w01 + v10 * w10 + v11 * w11;
    }
  }
  #pragma unroll
  for (int c = 0; c < 4; c++)
    out[((size_t)b * 128 + h * 32 + cv0 + c) * N1 + n] = acc[c];
}

extern "C" void kernel_launch(void* const* d_in, const int* in_sizes, int n_in,
                              void* d_out, int out_size, void* d_ws, size_t ws_size,
                              hipStream_t stream) {
  (void)in_sizes; (void)n_in; (void)out_size; (void)ws_size;
  const float* x     = (const float*)d_in[0];
  const float* ln_w  = (const float*)d_in[1];
  const float* ln_b  = (const float*)d_in[2];
  const float* wq    = (const float*)d_in[3];
  const float* wk    = (const float*)d_in[4];
  const float* dw_w  = (const float*)d_in[5];
  const float* oln_w = (const float*)d_in[6];
  const float* oln_b = (const float*)d_in[7];
  const float* off_w = (const float*)d_in[8];
  const float* off_b = (const float*)d_in[9];
  const float* rpb   = (const float*)d_in[10];

  char* ws = (char*)d_ws;
  _Float16* Wh  = (_Float16*)(ws + 0);          //  16 KB: f16 weights [64][128]
  _Float16* xnh = (_Float16*)(ws + 16384);      //   4 MB: f16 LN(x) [16384][128]
  float* qbuf   = (float*)(ws + 4210688);       //   2 MB: q (B,32,64,64)
  float* kbuf   = (float*)(ws + 6307840);       //   2 MB: k (B,32,64,64)
  float* qhi    = (float*)(ws + 8404992);       //   8 MB: q upsampled (B,32,128,128)
  float* tbuf   = (float*)(ws + 16793600);      //   8 MB: offset hidden (8,16,128,128)
  float* coords = (float*)(ws + 25182208);      // 9.4 MB: (8,9,16384,2) iy,ix
  float* out = (float*)d_out;

  k_wcvt   <<<32,   256, 0, stream>>>(wq, wk, Wh);
  k_ln     <<<2048, 256, 0, stream>>>(x, ln_w, ln_b, xnh);
  k_gemm   <<<128,  256, 0, stream>>>(xnh, Wh, qbuf, kbuf);
  k_upsample<<<8192,256, 0, stream>>>(qbuf, qhi);
  k_dwln   <<<512,  256, 0, stream>>>(qhi, dw_w, oln_w, oln_b, tbuf);
  k_offsets<<<512,  256, 0, stream>>>(tbuf, off_w, off_b, coords);
  k_attn   <<<8192, 256, 0, stream>>>(x, kbuf, qhi, coords, rpb, out);
}
